// RQBottleneck_21990232556241
// MI455X (gfx1250) — matrix operand order repaired and checked by reading, not compiled, not run on hardware
//
#include <hip/hip_runtime.h>

// ---------------------------------------------------------------------------
// RQ bottleneck for gfx1250 (MI455X), wave32, WMMA bf16 split-precision path.
// dist(t,k) = ||c_k||^2 - 2 * xn_t . c_k  (token norm constant per row, drops
// out of argmin). Dots via 3x v_wmma_f32_16x16x32_bf16 (hi*hi + hi*lo + lo*hi)
// for ~fp32 accuracy at bf16 matrix-pipe rate.  hi/lo stored as SEPARATE bf16
// planes so fragments are raw global_load_b128 -> v16bf bitcasts (no VALU
// unpacking feeding the matrix pipe).
// ---------------------------------------------------------------------------

typedef __attribute__((ext_vector_type(16))) __bf16        v16bf;
typedef __attribute__((ext_vector_type(8)))  float         v8f;
typedef __attribute__((ext_vector_type(8)))  unsigned int  v8u;

#define NTOK   32768      // 16 * 2048 tokens
#define DDIM   256
#define KCODES 8192
#define DEPTH  4
#define NELEM  8388608    // NTOK * DDIM

// ---- helpers ---------------------------------------------------------------

__device__ __forceinline__ void split_bf(float v, unsigned short& h, unsigned short& l) {
    __bf16 hb = (__bf16)v;                 // RNE to bf16
    float  hf = (float)hb;
    __bf16 lb = (__bf16)(v - hf);          // residual term
    h = __builtin_bit_cast(unsigned short, hb);
    l = __builtin_bit_cast(unsigned short, lb);
}

// Two 16-byte runs of 8 bf16 each -> one v16bf fragment (pure b128 loads).
__device__ __forceinline__ v16bf load_frag(const unsigned short* __restrict__ p0,
                                           const unsigned short* __restrict__ p1) {
    uint4 a = *(const uint4*)p0;
    uint4 b = *(const uint4*)p1;
    v8u w = (v8u){a.x, a.y, a.z, a.w, b.x, b.y, b.z, b.w};
    return __builtin_bit_cast(v16bf, w);
}

__device__ __forceinline__ uint4 pack_pairs(const unsigned short* s) {
    return make_uint4((unsigned int)s[0] | ((unsigned int)s[1] << 16),
                      (unsigned int)s[2] | ((unsigned int)s[3] << 16),
                      (unsigned int)s[4] | ((unsigned int)s[5] << 16),
                      (unsigned int)s[6] | ((unsigned int)s[7] << 16));
}

// ---- init: residual = x, agg(out quants) = 0 -------------------------------

__global__ void __launch_bounds__(256)
init_kernel(const float* __restrict__ x, float* __restrict__ residual,
            float* __restrict__ quants) {
    size_t i = (size_t)blockIdx.x * 256 + threadIdx.x;
    residual[i] = x[i];
    quants[i]   = 0.0f;
}

// ---- codebook pack: split-bf16 planes + fp32 row norms (one wave/code) -----

__global__ void __launch_bounds__(256)
cb_pack_kernel(const float* __restrict__ cb, unsigned short* __restrict__ cbH,
               unsigned short* __restrict__ cbL, float* __restrict__ cnorm) {
    int wave = (int)((blockIdx.x * 256 + threadIdx.x) >> 5);  // code row
    int lane = threadIdx.x & 31;
    const float* c = cb + (size_t)wave * DDIM + lane * 8;
    float4 v0 = *(const float4*)(c);
    float4 v1 = *(const float4*)(c + 4);
    float ss = v0.x * v0.x + v0.y * v0.y + v0.z * v0.z + v0.w * v0.w +
               v1.x * v1.x + v1.y * v1.y + v1.z * v1.z + v1.w * v1.w;
#pragma unroll
    for (int off = 16; off > 0; off >>= 1) ss += __shfl_xor(ss, off, 32);
    float vv[8] = {v0.x, v0.y, v0.z, v0.w, v1.x, v1.y, v1.z, v1.w};
    unsigned short h[8], l[8];
#pragma unroll
    for (int i = 0; i < 8; ++i) split_bf(vv[i], h[i], l[i]);
    size_t o = (size_t)wave * DDIM + lane * 8;
    *(uint4*)(cbH + o) = pack_pairs(h);
    *(uint4*)(cbL + o) = pack_pairs(l);
    if (lane == 0) cnorm[wave] = ss;
}

// ---- residual normalize + split-pack (one wave per token) ------------------

__global__ void __launch_bounds__(256)
xn_pack_kernel(const float* __restrict__ residual, unsigned short* __restrict__ xnH,
               unsigned short* __restrict__ xnL) {
    int wave = (int)((blockIdx.x * 256 + threadIdx.x) >> 5);  // token
    int lane = threadIdx.x & 31;
    const float* r = residual + (size_t)wave * DDIM + lane * 8;
    float4 v0 = *(const float4*)(r);
    float4 v1 = *(const float4*)(r + 4);
    float ss = v0.x * v0.x + v0.y * v0.y + v0.z * v0.z + v0.w * v0.w +
               v1.x * v1.x + v1.y * v1.y + v1.z * v1.z + v1.w * v1.w;
#pragma unroll
    for (int off = 16; off > 0; off >>= 1) ss += __shfl_xor(ss, off, 32);
    float inv = 1.0f / fmaxf(sqrtf(ss), 1e-12f);
    float vv[8] = {v0.x * inv, v0.y * inv, v0.z * inv, v0.w * inv,
                   v1.x * inv, v1.y * inv, v1.z * inv, v1.w * inv};
    unsigned short h[8], l[8];
#pragma unroll
    for (int i = 0; i < 8; ++i) split_bf(vv[i], h[i], l[i]);
    size_t o = (size_t)wave * DDIM + lane * 8;
    *(uint4*)(xnH + o) = pack_pairs(h);
    *(uint4*)(xnL + o) = pack_pairs(l);
}

// ---- WMMA GEMM + argmin: each wave owns 16 token rows, sweeps all codes ----

__global__ void __launch_bounds__(256)
gemm_argmin_kernel(const unsigned short* __restrict__ xnH,
                   const unsigned short* __restrict__ xnL,
                   const unsigned short* __restrict__ cbH,
                   const unsigned short* __restrict__ cbL,
                   const float* __restrict__ cnorm,
                   int* __restrict__ bestIdx) {
    const int lane    = threadIdx.x & 31;
    const int wave    = threadIdx.x >> 5;
    const int rowBase = (blockIdx.x * 8 + wave) * 16;
    const int col     = lane & 15;   // A: row; B: column; C/D: N
    const int xhalf   = lane >> 4;   // which K-half this lane carries

    float bd[8];
    int   bi[8];
#pragma unroll
    for (int j = 0; j < 8; ++j) { bd[j] = 3.4e38f; bi[j] = 0; }

    const size_t aOff = (size_t)(rowBase + col) * DDIM;
    const unsigned short* aH = xnH + aOff;
    const unsigned short* aL = xnL + aOff;

    for (int n0 = 0; n0 < KCODES; n0 += 64) {
        v8f acc[4];
#pragma unroll
        for (int s = 0; s < 4; ++s) acc[s] = (v8f){0, 0, 0, 0, 0, 0, 0, 0};

        for (int k0 = 0; k0 < DDIM; k0 += 32) {
            // A frag: 16 contiguous K starting at k0+half*16
            const int a0 = k0 + xhalf * 16;
            v16bf ah = load_frag(aH + a0, aH + a0 + 8);
            v16bf al = load_frag(aL + a0, aL + a0 + 8);
#pragma unroll
            for (int s = 0; s < 4; ++s) {
                // B frag: 16 contiguous K starting at k0+half*16, per column
                const size_t bOff = (size_t)(n0 + s * 16 + col) * DDIM + k0 + xhalf * 16;
                v16bf bh = load_frag(cbH + bOff, cbH + bOff + 8);
                v16bf bl = load_frag(cbL + bOff, cbL + bOff + 8);
                acc[s] = __builtin_amdgcn_wmma_f32_16x16x32_bf16(
                    false, ah, false, bh, (short)0, acc[s], false, false);
                acc[s] = __builtin_amdgcn_wmma_f32_16x16x32_bf16(
                    false, ah, false, bl, (short)0, acc[s], false, false);
                acc[s] = __builtin_amdgcn_wmma_f32_16x16x32_bf16(
                    false, al, false, bh, (short)0, acc[s], false, false);
            }
        }
#pragma unroll
        for (int s = 0; s < 4; ++s) {
            int   n  = n0 + s * 16 + col;
            float cn = cnorm[n];
#pragma unroll
            for (int j = 0; j < 8; ++j) {
                float dist = fmaf(-2.0f, acc[s][j], cn);   // row M = j + 8*xhalf
                if (dist < bd[j]) { bd[j] = dist; bi[j] = n; }
            }
        }
    }

    // min-reduce across the 16 lanes of each half (cols 0..15)
#pragma unroll
    for (int off = 1; off < 16; off <<= 1) {
#pragma unroll
        for (int j = 0; j < 8; ++j) {
            float od = __shfl_xor(bd[j], off, 32);
            int   oi = __shfl_xor(bi[j], off, 32);
            if (od < bd[j] || (od == bd[j] && oi < bi[j])) { bd[j] = od; bi[j] = oi; }
        }
    }
    if (col == 0) {
#pragma unroll
        for (int j = 0; j < 8; ++j)
            bestIdx[rowBase + xhalf * 8 + j] = bi[j];
    }
}

// ---- gather + residual/agg update + commitment partials (deterministic) ----

__global__ void __launch_bounds__(256)
update_kernel(const float* __restrict__ x, const float* __restrict__ cb_d,
              const int* __restrict__ bestIdx, float* __restrict__ residual,
              float* __restrict__ quants, float* __restrict__ codesOut,
              float* __restrict__ commitPart, int depth) {
    __shared__ float red[256];
    int token = blockIdx.x;
    int tid   = threadIdx.x;
    int idx   = bestIdx[token];
    size_t eo = (size_t)token * DDIM + tid;
    float c = cb_d[(size_t)idx * DDIM + tid];
    float r = residual[eo] - c;  residual[eo] = r;
    float a = quants[eo] + c;    quants[eo]   = a;   // agg snapshot after depth
    float diff = x[eo] - a;
    red[tid] = diff * diff;
    __syncthreads();
#pragma unroll
    for (int s = 128; s > 0; s >>= 1) {
        if (tid < s) red[tid] += red[tid + s];
        __syncthreads();
    }
    if (tid == 0) {
        commitPart[depth * NTOK + token] = red[0];
        codesOut[(size_t)token * DEPTH + depth] = (float)idx;
    }
}

// ---- commitment loss: fixed-order reduction of per-block partials ----------

__global__ void __launch_bounds__(256)
finalize_kernel(const float* __restrict__ commitPart, float* __restrict__ lossOut) {
    __shared__ float red[256];
    float s = 0.0f;
    for (int i = threadIdx.x; i < DEPTH * NTOK; i += 256) s += commitPart[i];
    red[threadIdx.x] = s;
    __syncthreads();
#pragma unroll
    for (int t = 128; t > 0; t >>= 1) {
        if (threadIdx.x < t) red[threadIdx.x] += red[threadIdx.x + t];
        __syncthreads();
    }
    if (threadIdx.x == 0)
        lossOut[0] = red[0] / ((float)DEPTH * (float)NELEM) * 0.25f;
}

// ---------------------------------------------------------------------------

extern "C" void kernel_launch(void* const* d_in, const int* in_sizes, int n_in,
                              void* d_out, int out_size, void* d_ws, size_t ws_size,
                              hipStream_t stream) {
    (void)in_sizes; (void)n_in; (void)out_size; (void)ws_size;

    const float* x         = (const float*)d_in[0];   // [16,2048,256]
    const float* codebooks = (const float*)d_in[1];   // [4,8192,256]

    float* out      = (float*)d_out;
    float* quants   = out;                 // [NELEM] = final aggregate
    float* lossOut  = out + NELEM;         // [1]
    float* codesOut = out + NELEM + 1;     // [NTOK*DEPTH] as floats

    char* ws = (char*)d_ws;
    float*          residual   = (float*)(ws);                        // 32 MB
    unsigned short* xnH        = (unsigned short*)(ws + 33554432);    // 16 MB
    unsigned short* xnL        = (unsigned short*)(ws + 50331648);    // 16 MB
    unsigned short* cbH        = (unsigned short*)(ws + 67108864);    //  4 MB
    unsigned short* cbL        = (unsigned short*)(ws + 71303168);    //  4 MB
    float*          cnorm      = (float*)(ws + 75497472);             // 32 KB
    int*            bestIdx    = (int*)(ws + 75530240);               // 128 KB
    float*          commitPart = (float*)(ws + 75661312);             // 512 KB

    init_kernel<<<NELEM / 256, 256, 0, stream>>>(x, residual, quants);

    for (int d = 0; d < DEPTH; ++d) {
        const float* cb = codebooks + (size_t)d * KCODES * DDIM;
        cb_pack_kernel<<<KCODES * 32 / 256, 256, 0, stream>>>(cb, cbH, cbL, cnorm);
        xn_pack_kernel<<<NTOK * 32 / 256, 256, 0, stream>>>(residual, xnH, xnL);
        gemm_argmin_kernel<<<NTOK / 16 / 8, 256, 0, stream>>>(xnH, xnL, cbH, cbL,
                                                              cnorm, bestIdx);
        update_kernel<<<NTOK, 256, 0, stream>>>(x, cb, bestIdx, residual,
                                                quants, codesOut, commitPart, d);
    }

    finalize_kernel<<<1, 256, 0, stream>>>(commitPart, lossOut);
}
